// SwarmSetEquivariantTorso_44976897524262
// MI455X (gfx1250) — matrix operand order repaired and checked by reading, not compiled, not run on hardware
//
#include <hip/hip_runtime.h>
#include <hip/hip_bf16.h>
#include <math.h>

#define DEVI __device__ __forceinline__

typedef __attribute__((ext_vector_type(16))) _Float16 v16h;
typedef __attribute__((ext_vector_type(8)))  _Float16 v8h;
typedef __attribute__((ext_vector_type(8)))  float    v8f;

// ---------------- problem constants ----------------
static constexpr int ROWS   = 32 * 2048;     // B*N
static constexpr int EGO    = 32;
static constexpr int SLOTS  = 5;
static constexpr int PSD    = 15;
static constexpr int PAIR_D = 10;
static constexpr int D_OBS  = 198;           // 32 + 75 + 10 + 81
static constexpr int PAIR_OFF = EGO + SLOTS * PSD;  // 107
static constexpr int DM = 64, MH = 128;
static constexpr int OUTD = MH + 6;          // 134
static constexpr int RT = 16;                // rows per workgroup
static constexpr int TT = RT * SLOTS;        // 80 token-rows per workgroup

// ---------------- f16 col-major weight arena in d_ws (element offsets) ----------
// layout: dst[n*Kpad + k], Kpad multiple of 32, offsets multiple of 16 (32B align)
enum : int {
  W_TOK = 0,                              // K12->32, N64   (2048)
  W_QW  = 2048,   W_KW = 6144,  W_VW = 10240, W_OW = 14336, // 64x64 each
  W_M1  = 18432,  // 64x128
  W_M2  = 26624,  // 128x128
  W_M3  = 43008,  // 128x64
  W_E1  = 51200,  // K32, 128
  W_E2  = 55296,  // 128x128
  W_E3  = 71680,  // 128x64
  W_L1  = 79872,  // 128x128
  W_L2  = 96256,  // 128x128
  W_PK  = 112640, W_PV = 116736, W_PO = 120832, // 64x64
  W_H1  = 124928, // 160x128
  W_H2  = 145408, // 128x128
  W_H3  = 161792, // 128x128
  W_END = 178176  // f16 elements (356,352 bytes)
};

// ---------------- helpers (branch-free fast transcendentals) ----------------
DEVI float fast_exp(float x) { return __expf(x); }           // v_exp_f32 path
DEVI float fast_rcp(float x) { return __builtin_amdgcn_rcpf(x); }

// tanh-approx GELU, branch-free: tanh(u) = sign(u)*(1-e)/(1+e), e = exp(-2|u|)
DEVI float gelu(float x) {
  float u = 0.7978845608028654f * (x + 0.044715f * x * x * x);
  float au = fabsf(u);
  float e  = fast_exp(-2.0f * au);
  float t  = (1.0f - e) * fast_rcp(1.0f + e);
  t = copysignf(t, u);
  return 0.5f * x * (1.0f + t);
}
DEVI float nan0(float x) { return (x - x == 0.0f) ? x : 0.0f; } // nan/inf -> 0

// A fragment: 16x32 f16, row-major in LDS, stride lda (multiple of 8).
// lanes 0-15: row m0+lane, K k0+{0..7,16..23}; lanes 16-31: K k0+{8..15,24..31}.
DEVI v16h load_a_frag(const _Float16* __restrict__ A, int lda, int m0, int k0) {
  const int lane = threadIdx.x & 31;
  const int m  = m0 + (lane & 15);
  const int hi = lane >> 4;
  const _Float16* p = A + m * lda + k0 + hi * 8;
  v8h lo = *(const v8h*)p;
  v8h up = *(const v8h*)(p + 16);
  v16h r;
#pragma unroll
  for (int i = 0; i < 8; ++i) { r[i] = lo[i]; r[i + 8] = up[i]; }
  return r;
}

// B fragment: 32x16 f16 from global col-major weights (ldw = Kpad, mult of 32).
// lane L: column n0+(L&15), K k0+(L>>4)*8+{0..7,16..23}.
DEVI v16h load_b_frag(const _Float16* __restrict__ W, int ldw, int n0, int k0) {
  const int lane = threadIdx.x & 31;
  const int n  = n0 + (lane & 15);
  const int kk = k0 + ((lane >> 4) << 3);
  const _Float16* q = W + n * ldw + kk;
  v8h lo = *(const v8h*)q;
  v8h up = *(const v8h*)(q + 16);
  v16h r;
#pragma unroll
  for (int i = 0; i < 8; ++i) { r[i] = lo[i]; r[i + 8] = up[i]; }
  return r;
}

enum { ST_F16 = 1, ST_F32 = 2, ST_GELU = 4, ST_ADD = 8, ST_MASK = 16 };

// Cooperative tiled GEMM: C[M,N] = act(A[M,K] * W[K,N] + bias) with store flags.
// A: LDS f16 row-major. W: global f16 col-major. 8 waves sweep 16x16 tiles.
template <int MODE>
DEVI void gemm(const _Float16* __restrict__ A, int lda,
               const _Float16* __restrict__ W, int ldw,
               const float* __restrict__ bias,
               int M, int N, int K,
               _Float16* C16, int ldc16,
               float* C32, int ldc32,
               const float* maskArr) {
  const int wave = threadIdx.x >> 5;
  const int nw   = blockDim.x >> 5;
  const int lane = threadIdx.x & 31;
  const int mt = M >> 4, nt = N >> 4;
  for (int t = wave; t < mt * nt; t += nw) {
    const int m0 = (t % mt) << 4;
    const int n0 = (t / mt) << 4;
    v8f acc = {};
    for (int k0 = 0; k0 < K; k0 += 32) {
      v16h a = load_a_frag(A, lda, m0, k0);
      v16h b = load_b_frag(W, ldw, n0, k0);
      acc = __builtin_amdgcn_wmma_f32_16x16x32_f16(false, a, false, b,
                                                   (short)0, acc, false, false);
    }
    const int n  = n0 + (lane & 15);
    const int mb = m0 + ((lane >> 4) << 3);
    const float bv = bias[n];
#pragma unroll
    for (int j = 0; j < 8; ++j) {
      const int m = mb + j;
      float v = acc[j] + bv;
      if (MODE & ST_GELU) v = gelu(v);
      if (MODE & ST_MASK) v *= maskArr[m];
      if (MODE & ST_F16)  C16[m * ldc16 + n] = (_Float16)v;
      if (MODE & ST_F32) {
        if (MODE & ST_ADD) C32[m * ldc32 + n] += v;
        else               C32[m * ldc32 + n]  = v;
      }
    }
  }
}

// ---------------- kernel params ----------------
struct TorsoParams {
  const float *obs;
  const float *tok_b, *ln1_s, *ln1_b;
  const float *a1_qb, *a1_kb, *a1_vb, *a1_ob;
  const float *ln2_s, *ln2_b;
  const float *m1_b, *m2_b, *m3_b;
  const float *e1_b, *e2_b, *e3_b;
  const float *l1_b, *l2_b, *l3_w, *l3_b;
  const float *seed, *p_qw, *p_qb, *p_kb, *p_vb, *p_ob;
  const float *pr_w, *pr_b;
  const float *h1_b, *h2_b, *h3_b;
  const _Float16 *W;
  float *out;
};

// ---------------- weight prep: f32 row-major (K,N) -> f16 col-major (N,Kpad) ---
__global__ void convert_w(const float* __restrict__ src, _Float16* __restrict__ dst,
                          int K, int N, int Kpad) {
  int i = blockIdx.x * blockDim.x + threadIdx.x;
  if (i >= N * Kpad) return;
  int n = i / Kpad, k = i - n * Kpad;
  dst[i] = (k < K) ? (_Float16)src[k * N + n] : (_Float16)0.0f;
}

// ---------------- main fused kernel: 16 rows per workgroup, 8 waves ------------
__global__ __launch_bounds__(256) void torso_kernel(TorsoParams p) {
  __shared__ float s_obs[RT * D_OBS];                    // staged obs rows
  __shared__ float s_tok[TT * DM];                       // f32 residual stream
  __shared__ __align__(32) _Float16 s_A [TT * 160];      // GEMM A staging
  __shared__ __align__(32) _Float16 s_H1[TT * MH];
  __shared__ __align__(32) _Float16 s_H2[TT * MH];
  __shared__ __align__(32) _Float16 s_K [TT * DM];
  __shared__ __align__(32) _Float16 s_V [TT * DM];
  __shared__ float s_mask[TT], s_msafe[TT], s_many[RT];
  __shared__ float s_logit[TT], s_qv[DM], s_c[RT * DM];

  const int tid  = threadIdx.x;
  const int row0 = blockIdx.x * RT;

  // warm L2/WGP$ with the first-layer weights while obs staging is in flight
  if (tid < 64) __builtin_prefetch(p.W + W_TOK + tid * 32, 0, 1);

  // ---- stage obs (16 rows x 198 f32) ----
  for (int i = tid; i < RT * D_OBS; i += 256)
    s_obs[i] = p.obs[(size_t)row0 * D_OBS + i];
  __syncthreads();

  // ---- masks per (row,slot) ----
  if (tid < TT) {
    int r = tid / SLOTS, s = tid - r * SLOTS;
    const float* rb = &s_obs[r * D_OBS + EGO + s * PSD];
    bool m = (fabsf(rb[0]) > 1e-6f) || (fabsf(rb[1]) > 1e-6f) || (fabsf(rb[2]) > 1e-6f);
    s_mask[tid] = m ? 1.0f : 0.0f;
  }
  __syncthreads();
  if (tid < RT) {
    float any = 0.0f;
    for (int s = 0; s < SLOTS; ++s) any = fmaxf(any, s_mask[tid * SLOTS + s]);
    s_many[tid] = any;
    for (int s = 0; s < SLOTS; ++s)
      s_msafe[tid * SLOTS + s] = (any > 0.5f) ? s_mask[tid * SLOTS + s]
                                              : (s == 0 ? 1.0f : 0.0f);
  }
  __syncthreads();

  // ---- token featurizer input (12 dims, zero-padded to K=32) ----
  if (tid < TT) {
    int r = tid / SLOTS, s = tid - r * SLOTS;
    const float* base = &s_obs[r * D_OBS + EGO];
    float rc[3];
    for (int c = 0; c < 3; ++c) rc[c] = nan0(base[s * PSD + c]);
    bool ms = s_mask[tid] > 0.5f;
    float dmin = 1e9f, dsum = 0.0f; int cnt = 0;
    for (int j = 0; j < SLOTS; ++j) {
      if (j == s) continue;
      if (!(ms && s_mask[r * SLOTS + j] > 0.5f)) continue;
      float dx = rc[0] - nan0(base[j * PSD + 0]);
      float dy = rc[1] - nan0(base[j * PSD + 1]);
      float dz = rc[2] - nan0(base[j * PSD + 2]);
      float d  = sqrtf(dx * dx + dy * dy + dz * dz);
      dmin = fminf(dmin, d); dsum += d; ++cnt;
    }
    float dmean = (cnt > 0) ? dsum * fast_rcp((float)cnt + 1e-9f) : 0.0f;
    float ax = base[s * PSD + 6], ay = base[s * PSD + 7], az = base[s * PSD + 8];
    float an = sqrtf(ax * ax + ay * ay + az * az);
    _Float16* w = &s_A[tid * 32];
    for (int c = 0; c < 4; ++c) w[c] = (_Float16)base[s * PSD + 11 + c]; // inv
    w[4] = (_Float16)base[s * PSD + 9];  // fre
    w[5] = (_Float16)base[s * PSD + 10]; // tem
    w[6] = (_Float16)ax; w[7] = (_Float16)ay; w[8] = (_Float16)az;
    w[9] = (_Float16)an; w[10] = (_Float16)dmin; w[11] = (_Float16)dmean;
    for (int c = 12; c < 32; ++c) w[c] = (_Float16)0.0f;
  }
  __syncthreads();

  // ---- tok = mask * gelu(tok_in @ tok_w + b) -> f32 residual ----
  gemm<ST_F32 | ST_GELU | ST_MASK>(s_A, 32, p.W + W_TOK, 32, p.tok_b,
                                   TT, DM, 32, nullptr, 0, s_tok, DM, s_mask);
  __syncthreads();

  // ---- LN1 -> s_A f16 ----
  if (tid < TT) {
    const float* x = &s_tok[tid * DM];
    float m = 0.f; for (int i = 0; i < DM; ++i) m += x[i]; m *= (1.0f / DM);
    float v = 0.f; for (int i = 0; i < DM; ++i) { float d = x[i] - m; v += d * d; }
    v *= (1.0f / DM);
    float rs = rsqrtf(v + 1e-6f);
    for (int i = 0; i < DM; ++i)
      s_A[tid * DM + i] = (_Float16)((x[i] - m) * rs * p.ln1_s[i] + p.ln1_b[i]);
  }
  __syncthreads();

  // ---- q,k,v projections (disjoint outputs, shared A) ----
  gemm<ST_F16>(s_A, DM, p.W + W_QW, 64, p.a1_qb, TT, DM, DM, s_H1, DM, nullptr, 0, nullptr);
  gemm<ST_F16>(s_A, DM, p.W + W_KW, 64, p.a1_kb, TT, DM, DM, s_K,  DM, nullptr, 0, nullptr);
  gemm<ST_F16>(s_A, DM, p.W + W_VW, 64, p.a1_vb, TT, DM, DM, s_V,  DM, nullptr, 0, nullptr);
  __syncthreads();

  // ---- 5x5 masked attention per (row, head), dh=16 ----
  if (tid < RT * 4) {
    int r = tid >> 2, h = tid & 3, tb = r * SLOTS, ho = h * 16;
    float lg[SLOTS][SLOTS];
    for (int i = 0; i < SLOTS; ++i)
      for (int j = 0; j < SLOTS; ++j) {
        if (s_msafe[tb + i] > 0.5f && s_msafe[tb + j] > 0.5f) {
          float a = 0.f;
          for (int d = 0; d < 16; ++d)
            a += (float)s_H1[(tb + i) * DM + ho + d] * (float)s_K[(tb + j) * DM + ho + d];
          lg[i][j] = a * 0.25f;
        } else lg[i][j] = -3.0e38f;
      }
    for (int i = 0; i < SLOTS; ++i) {
      float mx = lg[i][0];
      for (int j = 1; j < SLOTS; ++j) mx = fmaxf(mx, lg[i][j]);
      float a[SLOTS], sum = 0.f;
      for (int j = 0; j < SLOTS; ++j) { a[j] = fast_exp(lg[i][j] - mx); sum += a[j]; }
      float inv = fast_rcp(sum);
      for (int d = 0; d < 16; ++d) {
        float o = 0.f;
        for (int j = 0; j < SLOTS; ++j)
          o += a[j] * inv * (float)s_V[(tb + j) * DM + ho + d];
        s_H2[(tb + i) * DM + ho + d] = (_Float16)o;
      }
    }
  }
  __syncthreads();

  // ---- output projection, masked, add to residual ----
  gemm<ST_F32 | ST_ADD | ST_MASK>(s_H2, DM, p.W + W_OW, 64, p.a1_ob,
                                  TT, DM, DM, nullptr, 0, s_tok, DM, s_mask);
  __syncthreads();

  // ---- LN2 -> s_A, then MLP m (64->128->128->64), masked add ----
  if (tid < TT) {
    const float* x = &s_tok[tid * DM];
    float m = 0.f; for (int i = 0; i < DM; ++i) m += x[i]; m *= (1.0f / DM);
    float v = 0.f; for (int i = 0; i < DM; ++i) { float d = x[i] - m; v += d * d; }
    v *= (1.0f / DM);
    float rs = rsqrtf(v + 1e-6f);
    for (int i = 0; i < DM; ++i)
      s_A[tid * DM + i] = (_Float16)((x[i] - m) * rs * p.ln2_s[i] + p.ln2_b[i]);
  }
  __syncthreads();
  gemm<ST_F16 | ST_GELU>(s_A, DM, p.W + W_M1, 64, p.m1_b, TT, MH, DM, s_H1, MH, nullptr, 0, nullptr);
  __syncthreads();
  gemm<ST_F16 | ST_GELU>(s_H1, MH, p.W + W_M2, 128, p.m2_b, TT, MH, MH, s_H2, MH, nullptr, 0, nullptr);
  __syncthreads();
  gemm<ST_F32 | ST_ADD | ST_MASK>(s_H2, MH, p.W + W_M3, 128, p.m3_b,
                                  TT, DM, MH, nullptr, 0, s_tok, DM, s_mask);
  __syncthreads();

  // ---- tok -> f16 into s_A[:, 0:64] (stride 128 for the logits MLP) ----
  for (int i = tid; i < TT * DM; i += 256) {
    int t = i >> 6, c = i & 63;
    s_A[t * MH + c] = (_Float16)s_tok[i];
  }
  // ---- ego staging -> s_V[16][32] ----
  for (int i = tid; i < RT * EGO; i += 256) {
    int r = i >> 5, c = i & 31;
    s_V[r * EGO + c] = (_Float16)s_obs[r * D_OBS + c];
  }
  __syncthreads();

  // ---- ego MLP (32->128->128->64) ----
  gemm<ST_F16 | ST_GELU>(s_V, EGO, p.W + W_E1, 32, p.e1_b, RT, MH, EGO, s_H1, MH, nullptr, 0, nullptr);
  __syncthreads();
  gemm<ST_F16 | ST_GELU>(s_H1, MH, p.W + W_E2, 128, p.e2_b, RT, MH, MH, s_H2, MH, nullptr, 0, nullptr);
  __syncthreads();
  gemm<ST_F16>(s_H2, MH, p.W + W_E3, 128, p.e3_b, RT, DM, MH, s_K, DM, nullptr, 0, nullptr);
  __syncthreads();

  // ---- broadcast ego_ctx to 5 slots: s_A[:, 64:128] ----
  for (int i = tid; i < TT * DM; i += 256) {
    int t = i >> 6, c = i & 63;
    s_A[t * MH + DM + c] = s_K[(t / SLOTS) * DM + c];
  }
  __syncthreads();

  // ---- logits MLP (128->128->128->1) ----
  gemm<ST_F16 | ST_GELU>(s_A, MH, p.W + W_L1, 128, p.l1_b, TT, MH, MH, s_H1, MH, nullptr, 0, nullptr);
  __syncthreads();
  gemm<ST_F16 | ST_GELU>(s_H1, MH, p.W + W_L2, 128, p.l2_b, TT, MH, MH, s_H2, MH, nullptr, 0, nullptr);
  __syncthreads();
  if (tid < TT) {
    float a = p.l3_b[0];
    for (int i = 0; i < MH; ++i) a += (float)s_H2[tid * MH + i] * p.l3_w[i];
    s_logit[tid] = a;
  }
  // ---- pool query vector (constant across rows): q = seed @ p_qw + p_qb ----
  if (tid < DM) {
    float a = p.p_qb[tid];
    for (int k = 0; k < DM; ++k) a += p.seed[k] * p.p_qw[k * DM + tid];
    s_qv[tid] = a;
  }
  __syncthreads();

  // ---- alpha softmax + equivariant v_r / v_u outputs ----
  if (tid < RT) {
    float e[SLOTS], mx = -1e30f;
    for (int s = 0; s < SLOTS; ++s) {
      float ml = (s_mask[tid * SLOTS + s] > 0.5f) ? s_logit[tid * SLOTS + s] : -1e9f;
      e[s] = ml; mx = fmaxf(mx, ml);
    }
    float sum = 0.f;
    for (int s = 0; s < SLOTS; ++s) { e[s] = fast_exp(e[s] - mx) * s_mask[tid * SLOTS + s]; sum += e[s]; }
    float inv = fast_rcp(sum + 1e-9f);
    float vr[3] = {0, 0, 0}, vu[3] = {0, 0, 0};
    for (int s = 0; s < SLOTS; ++s) {
      float al = e[s] * inv;
      const float* rb = &s_obs[tid * D_OBS + EGO + s * PSD];
      for (int c = 0; c < 3; ++c) { vr[c] += al * rb[c]; vu[c] += al * rb[3 + c]; }
    }
    size_t o = (size_t)(row0 + tid) * OUTD;
    for (int c = 0; c < 3; ++c) { p.out[o + MH + c] = vr[c]; p.out[o + MH + 3 + c] = vu[c]; }
  }

  // ---- pool k,v projections from tok (A = s_A cols 0:64, stride 128) ----
  gemm<ST_F16>(s_A, MH, p.W + W_PK, 64, p.p_kb, TT, DM, DM, s_K, DM, nullptr, 0, nullptr);
  gemm<ST_F16>(s_A, MH, p.W + W_PV, 64, p.p_vb, TT, DM, DM, s_V, DM, nullptr, 0, nullptr);
  __syncthreads();

  // ---- pooled attention per (row, head) ----
  if (tid < RT * 4) {
    int r = tid >> 2, h = tid & 3, tb = r * SLOTS, ho = h * 16;
    float lg[SLOTS];
    for (int j = 0; j < SLOTS; ++j) {
      if (s_msafe[tb + j] > 0.5f) {
        float a = 0.f;
        for (int d = 0; d < 16; ++d)
          a += s_qv[ho + d] * (float)s_K[(tb + j) * DM + ho + d];
        lg[j] = a * 0.25f;
      } else lg[j] = -3.0e38f;
    }
    float mx = lg[0];
    for (int j = 1; j < SLOTS; ++j) mx = fmaxf(mx, lg[j]);
    float a[SLOTS], sum = 0.f;
    for (int j = 0; j < SLOTS; ++j) { a[j] = fast_exp(lg[j] - mx); sum += a[j]; }
    float inv = fast_rcp(sum);
    for (int d = 0; d < 16; ++d) {
      float c = 0.f;
      for (int j = 0; j < SLOTS; ++j)
        c += a[j] * inv * (float)s_V[(tb + j) * DM + ho + d];
      s_H1[r * DM + ho + d] = (_Float16)c;
    }
  }
  __syncthreads();

  // ---- pooled output projection -> c (f32) ----
  gemm<ST_F32>(s_H1, DM, p.W + W_PO, 64, p.p_ob, RT, DM, DM, nullptr, 0, s_c, DM, nullptr);
  __syncthreads();

  // ---- build h_in [16][160] = [ego | mask_any*c | gelu(pair@pr_w+b)] ----
  for (int i = tid; i < RT * EGO; i += 256) {
    int r = i >> 5, c = i & 31;
    s_A[r * 160 + c] = (_Float16)s_obs[r * D_OBS + c];
  }
  for (int i = tid; i < RT * DM; i += 256) {
    int r = i >> 6, c = i & 63;
    s_A[r * 160 + EGO + c] = (_Float16)(s_many[r] * s_c[r * DM + c]);
  }
  for (int i = tid; i < RT * DM; i += 256) {
    int r = i >> 6, n = i & 63;
    float a = p.pr_b[n];
    for (int k = 0; k < PAIR_D; ++k)
      a += s_obs[r * D_OBS + PAIR_OFF + k] * p.pr_w[k * DM + n];
    s_A[r * 160 + EGO + DM + n] = (_Float16)gelu(a);
  }
  __syncthreads();

  // ---- head MLP (160->128->128->128), final layer straight to global ----
  gemm<ST_F16 | ST_GELU>(s_A, 160, p.W + W_H1, 160, p.h1_b, RT, MH, 160, s_H1, MH, nullptr, 0, nullptr);
  __syncthreads();
  gemm<ST_F16 | ST_GELU>(s_H1, MH, p.W + W_H2, 128, p.h2_b, RT, MH, MH, s_H2, MH, nullptr, 0, nullptr);
  __syncthreads();
  gemm<ST_F32>(s_H2, MH, p.W + W_H3, 128, p.h3_b, RT, MH, MH,
               nullptr, 0, p.out + (size_t)row0 * OUTD, OUTD, nullptr);
}

// ---------------- launcher ----------------
extern "C" void kernel_launch(void* const* d_in, const int* in_sizes, int n_in,
                              void* d_out, int out_size, void* d_ws, size_t ws_size,
                              hipStream_t stream) {
  (void)in_sizes; (void)n_in; (void)out_size; (void)ws_size;
  _Float16* Wf16 = (_Float16*)d_ws;

  auto cw = [&](int idx, int off, int K, int N, int Kpad) {
    int total = N * Kpad;
    convert_w<<<(total + 255) / 256, 256, 0, stream>>>(
        (const float*)d_in[idx], Wf16 + off, K, N, Kpad);
  };
  cw(1,  W_TOK, 12, 64, 32);
  cw(5,  W_QW, 64, 64, 64);   cw(7,  W_KW, 64, 64, 64);
  cw(9,  W_VW, 64, 64, 64);   cw(11, W_OW, 64, 64, 64);
  cw(15, W_M1, 64, 128, 64);  cw(17, W_M2, 128, 128, 128); cw(19, W_M3, 128, 64, 128);
  cw(21, W_E1, 32, 128, 32);  cw(23, W_E2, 128, 128, 128); cw(25, W_E3, 128, 64, 128);
  cw(27, W_L1, 128, 128, 128); cw(29, W_L2, 128, 128, 128);
  cw(36, W_PK, 64, 64, 64);   cw(38, W_PV, 64, 64, 64);    cw(40, W_PO, 64, 64, 64);
  cw(44, W_H1, 160, 128, 160); cw(46, W_H2, 128, 128, 128); cw(48, W_H3, 128, 128, 128);

  TorsoParams p;
  p.obs   = (const float*)d_in[0];
  p.tok_b = (const float*)d_in[2];
  p.ln1_s = (const float*)d_in[3];  p.ln1_b = (const float*)d_in[4];
  p.a1_qb = (const float*)d_in[6];  p.a1_kb = (const float*)d_in[8];
  p.a1_vb = (const float*)d_in[10]; p.a1_ob = (const float*)d_in[12];
  p.ln2_s = (const float*)d_in[13]; p.ln2_b = (const float*)d_in[14];
  p.m1_b  = (const float*)d_in[16]; p.m2_b = (const float*)d_in[18]; p.m3_b = (const float*)d_in[20];
  p.e1_b  = (const float*)d_in[22]; p.e2_b = (const float*)d_in[24]; p.e3_b = (const float*)d_in[26];
  p.l1_b  = (const float*)d_in[28]; p.l2_b = (const float*)d_in[30];
  p.l3_w  = (const float*)d_in[31]; p.l3_b = (const float*)d_in[32];
  p.seed  = (const float*)d_in[33];
  p.p_qw  = (const float*)d_in[34]; p.p_qb = (const float*)d_in[35];
  p.p_kb  = (const float*)d_in[37]; p.p_vb = (const float*)d_in[39]; p.p_ob = (const float*)d_in[41];
  p.pr_w  = (const float*)d_in[42]; p.pr_b = (const float*)d_in[43];
  p.h1_b  = (const float*)d_in[45]; p.h2_b = (const float*)d_in[47]; p.h3_b = (const float*)d_in[49];
  p.W     = Wf16;
  p.out   = (float*)d_out;

  torso_kernel<<<ROWS / RT, 256, 0, stream>>>(p);
}
